// Level2GATEncoder_20117626814923
// MI455X (gfx1250) — matrix operand order repaired and checked by reading, not compiled, not run on hardware
//
#include <hip/hip_runtime.h>

#define NN 50000
#define EE 600000
#define HH 4
#define NTILES (NN / 16)   // 3125

typedef __attribute__((ext_vector_type(16))) __bf16 v16bf;
typedef __attribute__((ext_vector_type(8)))  float  v8f;
typedef __attribute__((ext_vector_type(4)))  unsigned u32x4;
typedef __attribute__((ext_vector_type(8)))  unsigned u32x8;

union AB16 { v16bf v; unsigned u[8]; uint4 q[2]; };

__device__ __forceinline__ unsigned short f2bf(float f) {
  unsigned u = __float_as_uint(f);
  unsigned r = (u + 0x7FFFu + ((u >> 16) & 1u)) >> 16;   // round-to-nearest-even
  return (unsigned short)r;
}

// monotone float<->uint encoding so segment-max can use deterministic integer atomics
__device__ __forceinline__ unsigned enc_f(float f) {
  unsigned u = __float_as_uint(f);
  return (u & 0x80000000u) ? ~u : (u | 0x80000000u);
}
__device__ __forceinline__ float dec_f(unsigned k) {
  unsigned u = (k & 0x80000000u) ? (k & 0x7FFFFFFFu) : ~k;
  return __uint_as_float(u);
}

// LDS byte offset of a __shared__ object (generic -> AS(3) -> int)
__device__ __forceinline__ unsigned lds_off_u32(const void* p) {
  return (unsigned)(size_t)(const __attribute__((address_space(3))) void*)p;
}

// Issue a TDM 1-D copy of `ndwords` dwords from global `gaddr` to LDS `lds_byte`.
// D# group0: count=1, lds_addr, global_addr[56:0], type=2 ("image").
// D# group1: data_size=4B, tensor_dim0=tile_dim0=ndwords, tensor_dim1=tile_dim1=1.
__device__ __forceinline__ void tdm_load_1d(unsigned lds_byte,
                                            unsigned long long gaddr,
                                            unsigned ndwords) {
  u32x4 g0;
  g0[0] = 1u;                                             // count=1 (valid user D#)
  g0[1] = lds_byte;                                       // lds_addr
  g0[2] = (unsigned)(gaddr & 0xFFFFFFFFu);                // global_addr[31:0]
  g0[3] = (unsigned)((gaddr >> 32) & 0x01FFFFFFu) | (2u << 30);  // [56:32] | type=2
  u32x8 g1;
  unsigned nd = ndwords;
  g1[0] = (2u << 16);                    // workgroup_mask=0, data_size=2 (4B)
  g1[1] = (nd & 0xFFFFu) << 16;          // tensor_dim0[15:0]
  g1[2] = ((nd >> 16) & 0xFFFFu) | (1u << 16);  // tensor_dim0[31:16], tensor_dim1=1
  g1[3] = (nd & 0xFFFFu) << 16;          // tensor_dim1 hi=0, tile_dim0=nd
  g1[4] = 1u;                            // tile_dim1=1, tile_dim2=0
  g1[5] = nd;                            // tensor_dim0_stride[31:0]
  g1[6] = (nd & 0xFFFFu) << 16;          // stride0 hi=0, tensor_dim1_stride lo
  g1[7] = 0u;
  asm volatile("tensor_load_to_lds %0, %1" :: "s"(g0), "s"(g1) : "memory");
}

// ---------------- f32 -> bf16 row-major conversion ----------------
__global__ void k_cvt_bf16(const float* __restrict__ in,
                           unsigned short* __restrict__ out, int n) {
  int i = blockIdx.x * blockDim.x + threadIdx.x;
  if (i < n) out[i] = f2bf(in[i]);
}

// ---------------- pack 128x128 f32 weight into WMMA B-fragments ----------------
// frag layout: [kt(4)][nt(8)][lane(32)][v(8)] dwords; dword v = {W[k][n], W[k+1][n]}
// B packing used here (16-bit, K=32, contiguous k order): lanes 0-15 hold K=0..15, lanes 16-31 hold K=16..31; N = lane&15
__global__ void k_pack_w(const float* __restrict__ W, unsigned* __restrict__ frag) {
  int tid = blockIdx.x * blockDim.x + threadIdx.x;
  if (tid >= 4 * 8 * 32 * 8) return;
  int v = tid & 7, lane = (tid >> 3) & 31, nt = (tid >> 8) & 7, kt = tid >> 11;
  int k = kt * 32 + 2 * v + ((lane & 16) ? 16 : 0);
  int n = nt * 16 + (lane & 15);
  unsigned lo = f2bf(W[k * 128 + n]);
  unsigned hi = f2bf(W[(k + 1) * 128 + n]);
  frag[tid] = lo | (hi << 16);
}

// ---------------- fused GEMM: xl = h@Wl + bl, xr = h@Wr + br ----------------
// 5 waves/block, one wave per 16-row tile; B fragments staged in LDS by the
// Tensor Data Mover, double-buffered so the kt+1 DMA overlaps the kt WMMAs.
// 64 v_wmma_f32_16x16x32_bf16 per wave.
__global__ __launch_bounds__(160) void k_gemm(
    const unsigned short* __restrict__ xb,
    const unsigned* __restrict__ fl, const unsigned* __restrict__ fr,
    const float* __restrict__ bl, const float* __restrict__ br,
    float* __restrict__ xl, float* __restrict__ xr) {
  __shared__ unsigned sB[2][2][2048];    // [buf][mat][fragment dwords], 32 KB

  int lane = threadIdx.x & 31;
  int tile = blockIdx.x * 5 + (threadIdx.x >> 5);   // grid is exactly 625 -> no guard
  int row0 = tile * 16;
  bool issuer = (threadIdx.x < 32);
  unsigned sb_base = lds_off_u32(&sB[0][0][0]);

  // preload kt=0 into buffer 0
  if (issuer) {
    tdm_load_1d(sb_base, (unsigned long long)(const void*)(fl), 2048);
    tdm_load_1d(sb_base + 8192, (unsigned long long)(const void*)(fr), 2048);
    __builtin_amdgcn_s_wait_tensorcnt(0);
  }
  __syncthreads();

  v8f zero = {0.f, 0.f, 0.f, 0.f, 0.f, 0.f, 0.f, 0.f};
  v8f accL[8], accR[8];
#pragma unroll
  for (int t = 0; t < 8; ++t) { accL[t] = zero; accR[t] = zero; }

  for (int kt = 0; kt < 4; ++kt) {
    int cur = kt & 1;
    // kick off DMA for next kt into the other buffer while we compute
    if (kt < 3 && issuer) {
      unsigned nb = sb_base + (unsigned)((cur ^ 1) * 16384);
      tdm_load_1d(nb, (unsigned long long)(const void*)(fl + (kt + 1) * 2048), 2048);
      tdm_load_1d(nb + 8192, (unsigned long long)(const void*)(fr + (kt + 1) * 2048), 2048);
    }

    // A fragment: lanes 0-15 hold K=0..15, lanes 16-31 hold K=16..31 (matches B packing)
    AB16 a;
    const unsigned short* xrow = xb + (size_t)(row0 + (lane & 15)) * 128 + kt * 32;
    int hiK = (lane & 16) ? 16 : 0;
#pragma unroll
    for (int v = 0; v < 8; ++v) {
      int k = 2 * v + hiK;
      a.u[v] = *(const unsigned*)(xrow + k);
    }
#pragma unroll
    for (int nt = 0; nt < 8; ++nt) {
      AB16 b;
      const uint4* pL = (const uint4*)&sB[cur][0][(nt * 32 + lane) * 8];
      b.q[0] = pL[0]; b.q[1] = pL[1];
      accL[nt] = __builtin_amdgcn_wmma_f32_16x16x32_bf16(
          false, a.v, false, b.v, (short)0, accL[nt], false, false);
      const uint4* pR = (const uint4*)&sB[cur][1][(nt * 32 + lane) * 8];
      b.q[0] = pR[0]; b.q[1] = pR[1];
      accR[nt] = __builtin_amdgcn_wmma_f32_16x16x32_bf16(
          false, a.v, false, b.v, (short)0, accR[nt], false, false);
    }

    if (kt < 3) {
      if (issuer) __builtin_amdgcn_s_wait_tensorcnt(0);
      __syncthreads();   // next buffer ready + everyone done reading current
    }
  }

  // C/D layout: VGPR v, lane: M = v + 8*(lane>=16), N = lane&15
  int rofs = (lane & 16) ? 8 : 0;
  int col0 = lane & 15;
#pragma unroll
  for (int nt = 0; nt < 8; ++nt) {
    int col = nt * 16 + col0;
    float bL = bl[col], bR = br[col];
#pragma unroll
    for (int v = 0; v < 8; ++v) {
      size_t o = (size_t)(row0 + v + rofs) * 128 + col;
      xl[o] = accL[nt][v] + bL;
      xr[o] = accR[nt][v] + bR;
    }
  }
}

// ---------------- edge attention score + deterministic segment max ----------------
__global__ void k_score(const float* __restrict__ xl, const float* __restrict__ xr,
                        const int* __restrict__ ei, const float* __restrict__ att,
                        float* __restrict__ score, unsigned* __restrict__ menc) {
  int idx = blockIdx.x * blockDim.x + threadIdx.x;
  if (idx >= EE * HH) return;
  int e = idx >> 2, h = idx & 3;
  int s = ei[e], d = ei[EE + e];
  const float4* pl = (const float4*)(xl + (size_t)s * 128 + h * 32);
  const float4* pr = (const float4*)(xr + (size_t)d * 128 + h * 32);
  const float4* pa = (const float4*)(att + h * 32);
  float sc = 0.f;
#pragma unroll
  for (int j = 0; j < 8; ++j) {
    float4 a = pl[j], b = pr[j], w = pa[j];
    float t;
    t = a.x + b.x; sc += w.x * (t > 0.f ? t : 0.2f * t);
    t = a.y + b.y; sc += w.y * (t > 0.f ? t : 0.2f * t);
    t = a.z + b.z; sc += w.z * (t > 0.f ? t : 0.2f * t);
    t = a.w + b.w; sc += w.w * (t > 0.f ? t : 0.2f * t);
  }
  score[idx] = sc;
  atomicMax(&menc[d * HH + h], enc_f(sc));
}

// ---------------- exp(score - m[dst]) + segment sum of denominators ----------------
__global__ void k_exp(float* __restrict__ score, const int* __restrict__ ei,
                      const unsigned* __restrict__ menc, float* __restrict__ den) {
  int idx = blockIdx.x * blockDim.x + threadIdx.x;
  if (idx >= EE * HH) return;
  int e = idx >> 2, h = idx & 3;
  int d = ei[EE + e];
  unsigned k = menc[d * HH + h];
  float m = k ? dec_f(k) : 0.f;
  float ex = __expf(score[idx] - m);
  score[idx] = ex;                 // reuse buffer for ex
  atomicAdd(&den[d * HH + h], ex);
}

// ---------------- weighted scatter-add aggregation ----------------
__global__ void k_agg(const float* __restrict__ ex, const int* __restrict__ ei,
                      const float* __restrict__ xl, const float* __restrict__ den,
                      float* __restrict__ agg) {
  int idx = blockIdx.x * blockDim.x + threadIdx.x;
  if (idx >= EE * HH) return;
  int e = idx >> 2, h = idx & 3;
  int s = ei[e], d = ei[EE + e];
  float alpha = ex[idx] / (den[d * HH + h] + 1e-16f);
  const float4* pl = (const float4*)(xl + (size_t)s * 128 + h * 32);
  float* po = agg + (size_t)d * 128 + h * 32;
#pragma unroll
  for (int j = 0; j < 8; ++j) {
    float4 a = pl[j];
    atomicAdd(po + j * 4 + 0, alpha * a.x);
    atomicAdd(po + j * 4 + 1, alpha * a.y);
    atomicAdd(po + j * 4 + 2, alpha * a.z);
    atomicAdd(po + j * 4 + 3, alpha * a.w);
  }
}

// ---------------- bias + LayerNorm + ELU, one wave per node ----------------
__global__ __launch_bounds__(256) void k_norm(const float* __restrict__ agg,
                                              const float* __restrict__ bias,
                                              const float* __restrict__ g,
                                              const float* __restrict__ bb,
                                              float* __restrict__ out) {
  int lane = threadIdx.x & 31;
  int node = blockIdx.x * 8 + (threadIdx.x >> 5);
  if (node >= NN) return;
  const float* row = agg + (size_t)node * 128;
  float v[4];
  float s = 0.f;
#pragma unroll
  for (int j = 0; j < 4; ++j) {
    int c = lane + j * 32;
    v[j] = row[c] + bias[c];
    s += v[j];
  }
#pragma unroll
  for (int o = 16; o > 0; o >>= 1) s += __shfl_xor(s, o, 32);
  float mu = s * (1.f / 128.f);
  float q = 0.f;
#pragma unroll
  for (int j = 0; j < 4; ++j) { float t = v[j] - mu; q += t * t; }
#pragma unroll
  for (int o = 16; o > 0; o >>= 1) q += __shfl_xor(q, o, 32);
  float rstd = rsqrtf(q * (1.f / 128.f) + 1e-5f);
#pragma unroll
  for (int j = 0; j < 4; ++j) {
    int c = lane + j * 32;
    float o2 = (v[j] - mu) * rstd * g[c] + bb[c];
    out[(size_t)node * 128 + c] = (o2 > 0.f) ? o2 : (__expf(o2) - 1.f);
  }
}

extern "C" void kernel_launch(void* const* d_in, const int* in_sizes, int n_in,
                              void* d_out, int out_size, void* d_ws, size_t ws_size,
                              hipStream_t stream) {
  (void)in_sizes; (void)n_in; (void)out_size; (void)ws_size;

  const float* x  = (const float*)d_in[0];
  const int*   ei = (const int*)d_in[1];
  const float* Wl[2]   = {(const float*)d_in[2],  (const float*)d_in[10]};
  const float* blv[2]  = {(const float*)d_in[3],  (const float*)d_in[11]};
  const float* Wr[2]   = {(const float*)d_in[4],  (const float*)d_in[12]};
  const float* brv[2]  = {(const float*)d_in[5],  (const float*)d_in[13]};
  const float* att[2]  = {(const float*)d_in[6],  (const float*)d_in[14]};
  const float* bias[2] = {(const float*)d_in[7],  (const float*)d_in[15]};
  const float* gam[2]  = {(const float*)d_in[8],  (const float*)d_in[16]};
  const float* bet[2]  = {(const float*)d_in[9],  (const float*)d_in[17]};

  char* ws = (char*)d_ws;
  size_t off = 0;
  auto carve = [&](size_t bytes) -> void* {
    off = (off + 255) & ~(size_t)255;
    void* p = ws + off;
    off += bytes;
    return p;
  };

  unsigned short* xb = (unsigned short*)carve((size_t)NN * 128 * 2);
  float* xl  = (float*)carve((size_t)NN * 128 * 4);
  float* xrr = (float*)carve((size_t)NN * 128 * 4);
  float* agg = (float*)carve((size_t)NN * 128 * 4);
  float* h1  = (float*)carve((size_t)NN * 128 * 4);
  float* sc  = (float*)carve((size_t)EE * HH * 4);
  unsigned* menc = (unsigned*)carve((size_t)NN * HH * 4);
  float* den = (float*)carve((size_t)NN * HH * 4);
  unsigned* wfragL[2], * wfragR[2];
  for (int L = 0; L < 2; ++L) {
    wfragL[L] = (unsigned*)carve(8192 * 4);
    wfragR[L] = (unsigned*)carve(8192 * 4);
  }

  // pack all 4 weight matrices into WMMA B-fragment layout (bf16)
  for (int L = 0; L < 2; ++L) {
    k_pack_w<<<32, 256, 0, stream>>>(Wl[L], wfragL[L]);
    k_pack_w<<<32, 256, 0, stream>>>(Wr[L], wfragR[L]);
  }

  const int eg = (EE * HH + 255) / 256;
  for (int L = 0; L < 2; ++L) {
    const float* hin = L ? h1 : x;
    float* hout = L ? (float*)d_out : h1;

    k_cvt_bf16<<<(NN * 128 + 255) / 256, 256, 0, stream>>>(hin, xb, NN * 128);
    hipMemsetAsync(agg, 0, (size_t)NN * 128 * 4, stream);
    hipMemsetAsync(menc, 0, (size_t)NN * HH * 4, stream);
    hipMemsetAsync(den, 0, (size_t)NN * HH * 4, stream);

    k_gemm<<<NTILES / 5, 160, 0, stream>>>(xb, wfragL[L], wfragR[L],
                                           blv[L], brv[L], xl, xrr);
    k_score<<<eg, 256, 0, stream>>>(xl, xrr, ei, att[L], sc, menc);
    k_exp<<<eg, 256, 0, stream>>>(sc, ei, menc, den);
    k_agg<<<eg, 256, 0, stream>>>(sc, ei, xl, den, agg);
    k_norm<<<(NN + 7) / 8, 256, 0, stream>>>(agg, bias[L], gam[L], bet[L], hout);
  }
}
